// SAGEEncoder_16819091931237
// MI455X (gfx1250) — matrix operand order repaired and checked by reading, not compiled, not run on hardware
//
#include <hip/hip_runtime.h>
#include <hip/hip_bf16.h>

typedef __attribute__((ext_vector_type(16))) _Float16 v16h;
typedef __attribute__((ext_vector_type(8)))  float    v8f;
typedef __attribute__((ext_vector_type(4)))  _Float16 v4h;

#define D128 128

// ---------------------------------------------------------------------------
// Weight prep: transpose to [N][K] f16, and fold Wk[1]*Watt[1] / Wv[1]*Wmsg[1]
// into combined 128x128 weights (kt = xp @ Wkatt replaces per-edge transform).
// ---------------------------------------------------------------------------
__global__ void prep_weights(const float* __restrict__ Wk, const float* __restrict__ Wq,
                             const float* __restrict__ Wv, const float* __restrict__ Watt,
                             const float* __restrict__ Wmsg, const float* __restrict__ Wout,
                             const float* __restrict__ linw,
                             _Float16* __restrict__ wq0t, _Float16* __restrict__ wkatt,
                             _Float16* __restrict__ wvmsg, _Float16* __restrict__ woutt,
                             _Float16* __restrict__ linwt) {
    int idx = blockIdx.x * blockDim.x + threadIdx.x;
    if (idx >= D128 * D128) return;
    int n = idx >> 7, k = idx & 127;
    wq0t[n * D128 + k]  = (_Float16)Wq[k * D128 + n];           // Wq[0]
    woutt[n * D128 + k] = (_Float16)Wout[k * D128 + n];         // Wout[0]
    linwt[n * D128 + k] = (_Float16)linw[k * D128 + n];
    // folded: Wkatt[k][h*32+f] = sum_c Wk[1][k][h*32+c] * Watt[1][h][c][f]
    int h = n >> 5, f = n & 31;
    const float* wk1 = Wk + D128 * D128;
    const float* wv1 = Wv + D128 * D128;
    const float* wa1 = Watt + 4 * 32 * 32;
    const float* wm1 = Wmsg + 4 * 32 * 32;
    float sk = 0.f, sv = 0.f;
#pragma unroll
    for (int c = 0; c < 32; ++c) {
        sk += wk1[k * D128 + h * 32 + c] * wa1[h * 1024 + c * 32 + f];
        sv += wv1[k * D128 + h * 32 + c] * wm1[h * 1024 + c * 32 + f];
    }
    wkatt[n * D128 + k] = (_Float16)sk;
    wvmsg[n * D128 + k] = (_Float16)sv;
}

// ---------------------------------------------------------------------------
// STEncoder: embedding gather, pad-masked mean over L=8, relu. Wave per node.
// ---------------------------------------------------------------------------
__global__ void encode_nodes(const int* __restrict__ tok, const float* __restrict__ emb,
                             float* __restrict__ xf, _Float16* __restrict__ xh, int N) {
    int wid  = (blockIdx.x * blockDim.x + threadIdx.x) >> 5;
    int lane = threadIdx.x & 31;
    if (wid >= N) return;
    const float4* emb4 = (const float4*)emb;
    float4 acc = {0.f, 0.f, 0.f, 0.f};
    float cnt = 0.f;
#pragma unroll
    for (int j = 0; j < 8; ++j) {
        int t = tok[wid * 8 + j];
        if (t != 0) {
            float4 e = emb4[(size_t)t * 32 + lane];
            acc.x += e.x; acc.y += e.y; acc.z += e.z; acc.w += e.w;
            cnt += 1.f;
        }
    }
    float inv = 1.f / fmaxf(cnt, 1.f);
    float4 r;
    r.x = fmaxf(acc.x * inv, 0.f); r.y = fmaxf(acc.y * inv, 0.f);
    r.z = fmaxf(acc.z * inv, 0.f); r.w = fmaxf(acc.w * inv, 0.f);
    if (xf) ((float4*)xf)[(size_t)wid * 32 + lane] = r;
    v4h h4 = { (_Float16)r.x, (_Float16)r.y, (_Float16)r.z, (_Float16)r.w };
    *(v4h*)(xh + (size_t)wid * D128 + lane * 4) = h4;
}

// ---------------------------------------------------------------------------
// Init accumulators (ws is poisoned; must re-init every launch).
// ---------------------------------------------------------------------------
__global__ void init_ws(float* __restrict__ aggu, float* __restrict__ dsum,
                        float* __restrict__ segmax, int nAgg, int nSeg) {
    int idx = blockIdx.x * blockDim.x + threadIdx.x;
    if (idx < nAgg) aggu[idx] = 0.f;
    if (idx < nSeg) { dsum[idx] = 0.f; segmax[idx] = -3.402823466e38f; }
}

// ---------------------------------------------------------------------------
// WMMA GEMM: C[M,128] = A[M,128](f16) @ B (Bt given as [N][K] f16)
//  + compile-time optional bias[N] / f32 skip; f32 and/or f16 outputs.
// Block = 64 rows x 128 cols, 256 threads: 8 waves own one 16-col tile each
// across 4 row subtiles -> 16 x v_wmma_f32_16x16x32_f16 per wave.
// Bt (32KB) is staged into LDS with GLOBAL_LOAD_ASYNC_TO_LDS_B128 (ASYNCcnt),
// A tile (16KB) with plain clamped loads. A/B fragments use contiguous k = 16*half + i.
// ---------------------------------------------------------------------------
template <bool HAS_SKIP, bool HAS_BIAS, bool OUT_F, bool OUT_H>
__global__ void __launch_bounds__(256)
gemm128(const _Float16* __restrict__ A, const _Float16* __restrict__ Bt,
        const float* __restrict__ skip, const float* __restrict__ bias,
        float* __restrict__ outF, _Float16* __restrict__ outH, int M) {
    __shared__ _Float16 lA[64 * D128];     // 16 KB
    __shared__ _Float16 lB[D128 * D128];   // 32 KB
    int t = threadIdx.x;
    int blockRow = blockIdx.x * 64;

    { // async-stage whole Bt (2048 x 16B), tracked by ASYNCcnt
        auto lBlds = (__attribute__((address_space(3))) char*)(void*)lB;
        unsigned ldsBase = (unsigned)(unsigned long long)lBlds;
        const char* gB = (const char*)Bt;
#pragma unroll
        for (int i = 0; i < 8; ++i) {
            unsigned chunk = (unsigned)(t + i * 256);
            unsigned laddr = ldsBase + chunk * 16u;
            const void* gaddr = gB + (size_t)chunk * 16u;
            asm volatile("global_load_async_to_lds_b128 %0, %1, off"
                         :: "v"(laddr), "v"(gaddr) : "memory");
        }
    }
    { // stage A tile (64x128 f16): 1024 chunks of 8 halves, row-clamped
#pragma unroll
        for (int i = 0; i < 4; ++i) {
            int chunk = t + i * 256;
            int row = chunk >> 4, c8 = (chunk & 15) * 8;
            int gr = blockRow + row; if (gr >= M) gr = M - 1;
            *(float4*)&lA[row * D128 + c8] = *(const float4*)&A[(size_t)gr * D128 + c8];
        }
    }
    asm volatile("s_wait_asynccnt 0" ::: "memory");
    __syncthreads();

    int lane = t & 31, wave = t >> 5;
    int half = lane >> 4, r = lane & 15, colTile = wave * 16;

    // preload the 4 B fragments for this wave's column tile
    // B 32x16 f16: lanes 0-15 col n hold K 0..15; lanes 16-31 hold K 16..31
    v16h bf[4];
#pragma unroll
    for (int kk = 0; kk < 4; ++kk) {
        const _Float16* pb = &lB[(colTile + r) * D128 + kk * 32 + half * 16];
#pragma unroll
        for (int i = 0; i < 16; ++i) bf[kk][i] = pb[i];
    }

    v8f acc[4] = {};
#pragma unroll
    for (int rt = 0; rt < 4; ++rt) {
#pragma unroll
        for (int kk = 0; kk < 4; ++kk) {
            // A 16x32 f16: lanes 0-15 row r hold K 0..15; lanes 16-31: 16..31
            v16h a;
            const _Float16* pa = &lA[(rt * 16 + r) * D128 + kk * 32 + half * 16];
#pragma unroll
            for (int i = 0; i < 8; ++i) { a[i] = pa[i]; a[i + 8] = pa[8 + i]; }
            acc[rt] = __builtin_amdgcn_wmma_f32_16x16x32_f16(false, a, false, bf[kk],
                                                             (short)0, acc[rt], false, false);
        }
    }

    // C layout: VGPR j -> (M=j, N=lane) lanes 0-15 ; (M=j+8, N=lane-16) lanes 16-31
    int col = colTile + r;
    float bv = HAS_BIAS ? bias[col] : 0.f;
#pragma unroll
    for (int rt = 0; rt < 4; ++rt) {
#pragma unroll
        for (int j = 0; j < 8; ++j) {
            int row = blockRow + rt * 16 + j + half * 8;
            if (row >= M) continue;
            float v = acc[rt][j] + bv;
            if (HAS_SKIP) v += skip[(size_t)row * D128 + col];
            if (OUT_F) outF[(size_t)row * D128 + col] = v;
            if (OUT_H) outH[(size_t)row * D128 + col] = (_Float16)v;
        }
    }
}

// ordered-int float atomic max (segmax initialized to -FLT_MAX)
__device__ __forceinline__ void atomicMaxF(float* addr, float v) {
    int iv = __float_as_int(v);
    if (iv >= 0) atomicMax((int*)addr, iv);
    else         atomicMin((unsigned int*)addr, (unsigned int)iv);
}

// ---------------------------------------------------------------------------
// Edge pass 1: per-edge per-head dot(kt[src], qa[dst]) -> score, seg atomic max
// Wave per edge: lane owns 4 dims, 8-lane tree reduce per head.
// ---------------------------------------------------------------------------
__global__ void edge_score(const int* __restrict__ edges, const float* __restrict__ kt,
                           const float* __restrict__ qa, const float* __restrict__ mu,
                           float* __restrict__ score, float* __restrict__ segmax, int E) {
    int e = (blockIdx.x * blockDim.x + threadIdx.x) >> 5;
    int lane = threadIdx.x & 31;
    if (e >= E) return;
    int src = edges[e], dst = edges[E + e];
    float4 a = ((const float4*)kt)[(size_t)src * 32 + lane];
    float4 b = ((const float4*)qa)[(size_t)dst * 32 + lane];
    float p = a.x * b.x + a.y * b.y + a.z * b.z + a.w * b.w;
    p += __shfl_xor(p, 1, 32);
    p += __shfl_xor(p, 2, 32);
    p += __shfl_xor(p, 4, 32);
    int h = lane >> 3;
    if ((lane & 7) == 0) {
        float s = p * mu[4 + h] * 0.17677669529663687f;  // mu[1][h] / sqrt(32)
        score[(size_t)e * 4 + h] = s;
        atomicMaxF(&segmax[(size_t)dst * 4 + h], s);
    }
}

// ---------------------------------------------------------------------------
// Edge pass 2: w = exp(s - segmax[dst]); scatter w and w*mt[src].
// ---------------------------------------------------------------------------
__global__ void edge_agg(const int* __restrict__ edges, const float* __restrict__ mt,
                         const float* __restrict__ score, const float* __restrict__ segmax,
                         float* __restrict__ dsum, float* __restrict__ aggu, int E) {
    int e = (blockIdx.x * blockDim.x + threadIdx.x) >> 5;
    int lane = threadIdx.x & 31;
    if (e >= E) return;
    int src = edges[e], dst = edges[E + e];
    int h = lane >> 3;
    float w = __expf(score[(size_t)e * 4 + h] - segmax[(size_t)dst * 4 + h]);
    if ((lane & 7) == 0) atomicAdd(&dsum[(size_t)dst * 4 + h], w);
    float4 mv = ((const float4*)mt)[(size_t)src * 32 + lane];
    float* d = &aggu[(size_t)dst * D128 + lane * 4];
    atomicAdd(d + 0, w * mv.x);
    atomicAdd(d + 1, w * mv.y);
    atomicAdd(d + 2, w * mv.z);
    atomicAdd(d + 3, w * mv.w);
}

// ---------------------------------------------------------------------------
// Finalize: normalize by denominator, tanh-approx gelu, emit f16 GEMM input.
// ---------------------------------------------------------------------------
__global__ void finalize_gelu(const float* __restrict__ aggu, const float* __restrict__ dsum,
                              _Float16* __restrict__ g16, int total) {
    int idx = blockIdx.x * blockDim.x + threadIdx.x;
    if (idx >= total) return;
    int n = idx >> 7, c = idx & 127, h = c >> 5;
    float v = aggu[idx] / (dsum[n * 4 + h] + 1e-9f);
    float g = 0.5f * v * (1.f + tanhf(0.7978845608028654f * (v + 0.044715f * v * v * v)));
    g16[idx] = (_Float16)g;
}

// ---------------------------------------------------------------------------
extern "C" void kernel_launch(void* const* d_in, const int* in_sizes, int n_in,
                              void* d_out, int out_size, void* d_ws, size_t ws_size,
                              hipStream_t stream) {
    const int*   tokA    = (const int*)d_in[0];
    const int*   tokP    = (const int*)d_in[1];
    // d_in[2] (edge_ap) is dead code in the reference output path
    const int*   edge_pa = (const int*)d_in[3];
    const float* emb     = (const float*)d_in[4];
    const float* Wk      = (const float*)d_in[5];
    const float* Wq      = (const float*)d_in[6];
    const float* Wv      = (const float*)d_in[7];
    const float* Watt    = (const float*)d_in[8];
    const float* Wmsg    = (const float*)d_in[9];
    const float* mu      = (const float*)d_in[10];
    const float* Wout    = (const float*)d_in[11];
    const float* linw    = (const float*)d_in[12];
    const float* linb    = (const float*)d_in[13];
    float* out = (float*)d_out;

    int Na = in_sizes[0] / 8;
    int Np = in_sizes[1] / 8;
    int E  = in_sizes[3] / 2;

    char* p = (char*)d_ws;
    auto alloc = [&](size_t bytes) {
        char* r = p; p += (bytes + 255) & ~(size_t)255; return r;
    };
    float*    xa    = (float*)   alloc((size_t)Na * D128 * 4);
    _Float16* xa16  = (_Float16*)alloc((size_t)Na * D128 * 2);
    _Float16* xp16  = (_Float16*)alloc((size_t)Np * D128 * 2);
    float*    qa    = (float*)   alloc((size_t)Na * D128 * 4);
    float*    kt    = (float*)   alloc((size_t)Np * D128 * 4);
    float*    mt    = (float*)   alloc((size_t)Np * D128 * 4);
    float*    score = (float*)   alloc((size_t)E * 4 * 4);
    float*    segmx = (float*)   alloc((size_t)Na * 4 * 4);
    float*    dsum  = (float*)   alloc((size_t)Na * 4 * 4);
    float*    aggu  = (float*)   alloc((size_t)Na * D128 * 4);
    _Float16* g16   = (_Float16*)alloc((size_t)Na * D128 * 2);
    _Float16* h16   = (_Float16*)alloc((size_t)Na * D128 * 2);
    _Float16* wq0t  = (_Float16*)alloc(D128 * D128 * 2);
    _Float16* wkatt = (_Float16*)alloc(D128 * D128 * 2);
    _Float16* wvmsg = (_Float16*)alloc(D128 * D128 * 2);
    _Float16* woutt = (_Float16*)alloc(D128 * D128 * 2);
    _Float16* linwt = (_Float16*)alloc(D128 * D128 * 2);

    prep_weights<<<64, 256, 0, stream>>>(Wk, Wq, Wv, Watt, Wmsg, Wout, linw,
                                         wq0t, wkatt, wvmsg, woutt, linwt);
    encode_nodes<<<(Na + 7) / 8, 256, 0, stream>>>(tokA, emb, xa, xa16, Na);
    encode_nodes<<<(Np + 7) / 8, 256, 0, stream>>>(tokP, emb, nullptr, xp16, Np);
    init_ws<<<(Na * D128 + 255) / 256, 256, 0, stream>>>(aggu, dsum, segmx,
                                                         Na * D128, Na * 4);
    gemm128<false, false, true, false><<<(Na + 63) / 64, 256, 0, stream>>>(
        xa16, wq0t, nullptr, nullptr, qa, nullptr, Na);
    gemm128<false, false, true, false><<<(Np + 63) / 64, 256, 0, stream>>>(
        xp16, wkatt, nullptr, nullptr, kt, nullptr, Np);
    gemm128<false, false, true, false><<<(Np + 63) / 64, 256, 0, stream>>>(
        xp16, wvmsg, nullptr, nullptr, mt, nullptr, Np);
    edge_score<<<(E + 7) / 8, 256, 0, stream>>>(edge_pa, kt, qa, mu, score, segmx, E);
    edge_agg<<<(E + 7) / 8, 256, 0, stream>>>(edge_pa, mt, score, segmx, dsum, aggu, E);
    finalize_gelu<<<(Na * D128 + 255) / 256, 256, 0, stream>>>(aggu, dsum, g16, Na * D128);
    gemm128<true, false, false, true><<<(Na + 63) / 64, 256, 0, stream>>>(
        g16, woutt, xa, nullptr, nullptr, h16, Na);
    gemm128<false, true, true, false><<<(Na + 63) / 64, 256, 0, stream>>>(
        h16, linwt, nullptr, linb, out, nullptr, Na);
}
